// MicroExpert_53936199303680
// MI455X (gfx1250) — matrix operand order repaired and checked by reading, not compiled, not run on hardware
//
#include <hip/hip_runtime.h>

#define TT   2048
#define DD   512
#define NH   8
#define HDIM 64
#define HWIN 256
#define MROWS (2 * TT)   // B*T = 4096

typedef __attribute__((ext_vector_type(16))) _Float16 v16h;
typedef __attribute__((ext_vector_type(8)))  float    v8f;

union Frag { v16h v; unsigned u[8]; };

__device__ __forceinline__ v8f wmma16(v16h a, v16h b, v8f c) {
  return __builtin_amdgcn_wmma_f32_16x16x32_f16(false, a, false, b, (short)0, c,
                                                false, false);
}

// convert float4 -> 4 halves, store as two dword LDS writes (dst 4B-aligned)
__device__ __forceinline__ void cvt_store4(_Float16* dst, float4 f) {
  union { _Float16 h[4]; unsigned u[2]; } p;
  p.h[0] = (_Float16)f.x; p.h[1] = (_Float16)f.y;
  p.h[2] = (_Float16)f.z; p.h[3] = (_Float16)f.w;
  unsigned* d = (unsigned*)dst;
  d[0] = p.u[0]; d[1] = p.u[1];
}

// A-matrix (16x32 f16) fragment per ISA layout:
// lane row = lane&15 ; halves pair p -> K = 8*(lane>>4) + (p<4 ? 2p : 2p+8)
__device__ __forceinline__ v16h load_afrag(const _Float16* tile, int ld, int lo, int hi) {
  Frag f;
  const _Float16* rp = tile + lo * ld;
#pragma unroll
  for (int p = 0; p < 8; ++p) {
    int kk = 8 * hi + ((p < 4) ? (2 * p) : (2 * p + 8));
    f.u[p] = *(const unsigned*)(rp + kk);
  }
  return f.v;
}

// B-matrix (32x16 f16) fragment: lane col = lane&15 ; pair p -> K = 8*(lane>>4) + (p<4 ? 2p : 2p+8)
// tile stored transposed in LDS as [n][k] so pairs are contiguous
__device__ __forceinline__ v16h load_bfrag(const _Float16* tile, int ld, int lo, int hi) {
  Frag f;
  const _Float16* rp = tile + lo * ld + 8 * hi;
#pragma unroll
  for (int p = 0; p < 8; ++p) f.u[p] = *(const unsigned*)(rp + ((p < 4) ? (2 * p) : (2 * p + 8)));
  return f.v;
}

// ---------------------------------------------------------------------------
// Generic GEMM: C[M,N] = A[M,K] * Bw[N,K]^T + bias (+resid) (ReLU optional)
// Tile 128x128, 8 waves of 32x64 (2 A-frags x 4 B-frags = 8 WMMA / k-step).
// taps==3: conv mode -> A rows shifted by (tap-1) with batch-boundary masking
//          done branchlessly (clamped row + 0/1 multiplier), Bw = [3][N][K].
// Requires: M % 128 == 0, N % 128 == 0, K % 32 == 0 (true for all launches).
// ---------------------------------------------------------------------------
__global__ __launch_bounds__(256) void gemm_wmma(
    const float* __restrict__ A, const float* __restrict__ Bw,
    const float* __restrict__ bias, const float* __restrict__ resid,
    float* __restrict__ C, int M, int N, int K, int relu, int taps) {
  __shared__ _Float16 As[128 * 40];
  __shared__ _Float16 Bs[128 * 40];

  const int tid  = threadIdx.x;
  const int wid  = tid >> 5;
  const int lane = tid & 31;
  const int lo   = lane & 15;
  const int hi   = lane >> 4;
  const int m_off = (wid & 3) * 32;
  const int n_off = (wid >> 2) * 64;
  const int blockM = blockIdx.y * 128;
  const int blockN = blockIdx.x * 128;

  v8f acc[2][4] = {};

  const int srow = tid >> 1;            // staged row 0..127 (A and B)
  const int scg  = (tid & 1) * 16;      // col group 0 / 16

  for (int tap = 0; tap < taps; ++tap) {
    // per-tap, k-invariant row setup (branchless masking for conv shifts)
    const int gr = blockM + srow;
    float amask = 1.f;
    long rowi = gr;
    if (taps == 3) {
      const int ts = (gr & (TT - 1)) + tap - 1;
      const bool valid = (ts >= 0) && (ts < TT);
      amask = valid ? 1.f : 0.f;
      rowi = valid ? (long)(gr + tap - 1) : (long)gr;  // clamped, in-bounds
    }
    const float* arowp = A + rowi * (long)K;
    const float* browp = Bw + (size_t)tap * N * K + (size_t)(blockN + srow) * K;

    for (int k0 = 0; k0 < K; k0 += 32) {
      // --- stage A tile (f32 -> f16, masked) ---
      {
        const float* ap = arowp + k0 + scg;
        _Float16* dst = &As[srow * 40 + scg];
#pragma unroll
        for (int i = 0; i < 4; ++i) {
          float4 f = ((const float4*)ap)[i];
          f.x *= amask; f.y *= amask; f.z *= amask; f.w *= amask;
          cvt_store4(dst + i * 4, f);
        }
        if (k0 + 32 < K) __builtin_prefetch(ap + 32, 0, 1);
      }
      // --- stage B tile (weight rows, [n][k]) ---
      {
        const float* bp = browp + k0 + scg;
        _Float16* dst = &Bs[srow * 40 + scg];
#pragma unroll
        for (int i = 0; i < 4; ++i) cvt_store4(dst + i * 4, ((const float4*)bp)[i]);
        if (k0 + 32 < K) __builtin_prefetch(bp + 32, 0, 1);
      }
      __syncthreads();

      v16h af[2], bf[4];
#pragma unroll
      for (int mt = 0; mt < 2; ++mt)
        af[mt] = load_afrag(&As[(m_off + mt * 16) * 40], 40, lo, hi);
#pragma unroll
      for (int nt = 0; nt < 4; ++nt)
        bf[nt] = load_bfrag(&Bs[(n_off + nt * 16) * 40], 40, lo, hi);
#pragma unroll
      for (int mt = 0; mt < 2; ++mt)
#pragma unroll
        for (int nt = 0; nt < 4; ++nt)
          acc[mt][nt] = wmma16(af[mt], bf[nt], acc[mt][nt]);
      __syncthreads();
    }
  }

  // --- epilogue: bias / relu / residual (M,N exact multiples of tile) ---
#pragma unroll
  for (int mt = 0; mt < 2; ++mt) {
#pragma unroll
    for (int nt = 0; nt < 4; ++nt) {
#pragma unroll
      for (int r = 0; r < 8; ++r) {
        const int grow = blockM + m_off + mt * 16 + r + 8 * hi;
        const int gcol = blockN + n_off + nt * 16 + lo;
        float v = acc[mt][nt][r] + bias[gcol];
        if (relu) v = fmaxf(v, 0.f);
        const size_t off = (size_t)grow * N + gcol;
        if (resid) v += resid[off];
        C[off] = v;
      }
    }
  }
}

// ---------------------------------------------------------------------------
// Banded flash attention: workgroup = (b, h, 128 q rows), wave = 16-row strip.
// Q,K,V are [B*T, D] f32 with head slice at h*HDIM. HD = 64 -> 2 K-steps.
// ---------------------------------------------------------------------------
__global__ __launch_bounds__(256) void attn_wmma(
    const float* __restrict__ Q, const float* __restrict__ Kb,
    const float* __restrict__ Vb, float* __restrict__ ctx) {
  __shared__ _Float16 Klds[64 * 72];        // [key][hd]
  __shared__ _Float16 Vlds[64 * 72];        // [hd][key] (transposed)
  __shared__ _Float16 Plds[8 * 16 * 72];    // per-wave P strip

  const int tid  = threadIdx.x;
  const int wid  = tid >> 5;
  const int lane = tid & 31;
  const int lo   = lane & 15;
  const int hi   = lane >> 4;
  const int q0   = blockIdx.x * 128;
  const int b    = blockIdx.y / NH;
  const int h    = blockIdx.y % NH;

  // Q fragments, scaled by 1/sqrt(HD) = 1/8, kept resident
  v16h aq[2];
  {
    const float* qrow = Q + ((size_t)(b * TT + q0 + wid * 16 + lo)) * DD + h * HDIM;
#pragma unroll
    for (int st = 0; st < 2; ++st) {
      Frag f;
#pragma unroll
      for (int p = 0; p < 8; ++p) {
        int kk = st * 32 + 8 * hi + ((p < 4) ? 2 * p : 2 * p + 8);
        float2 q2 = *(const float2*)(qrow + kk);
        union { _Float16 h2[2]; unsigned u; } pk;
        pk.h2[0] = (_Float16)(q2.x * 0.125f);
        pk.h2[1] = (_Float16)(q2.y * 0.125f);
        f.u[p] = pk.u;
      }
      aq[st] = f.v;
    }
  }

  v8f acc[4] = {};
  float mrow[8], lrow[8];
#pragma unroll
  for (int r = 0; r < 8; ++r) { mrow[r] = -1e30f; lrow[r] = 0.f; }

  int kcs = q0 - HWIN; if (kcs < 0) kcs = 0;
  int kce = q0 + 128 + HWIN; if (kce > TT) kce = TT;

  const int srow = tid >> 2;          // staged key row 0..63
  const int scg  = (tid & 3) * 16;    // hd col group

  for (int kc = kcs; kc < kce; kc += 64) {
    // stage K chunk [key][hd]
    {
      const float* kp = Kb + ((size_t)(b * TT + kc + srow)) * DD + h * HDIM + scg;
      _Float16* dst = &Klds[srow * 72 + scg];
#pragma unroll
      for (int i = 0; i < 4; ++i) cvt_store4(dst + i * 4, ((const float4*)kp)[i]);
    }
    // stage V chunk transposed [hd][key]
    {
      const float* vp = Vb + ((size_t)(b * TT + kc + srow)) * DD + h * HDIM + scg;
#pragma unroll
      for (int i = 0; i < 4; ++i) {
        float4 f = ((const float4*)vp)[i];
        Vlds[(scg + i * 4 + 0) * 72 + srow] = (_Float16)f.x;
        Vlds[(scg + i * 4 + 1) * 72 + srow] = (_Float16)f.y;
        Vlds[(scg + i * 4 + 2) * 72 + srow] = (_Float16)f.z;
        Vlds[(scg + i * 4 + 3) * 72 + srow] = (_Float16)f.w;
      }
    }
    __syncthreads();

    // S = Q * K^T (strip 16 x 64)
    v8f s[4] = {};
#pragma unroll
    for (int nt = 0; nt < 4; ++nt) {
#pragma unroll
      for (int st = 0; st < 2; ++st) {
        v16h bf = load_bfrag(&Klds[(nt * 16) * 72 + st * 32], 72, lo, hi);
        s[nt] = wmma16(aq[st], bf, s[nt]);
      }
    }

    // band mask + online softmax (row = 16 lanes sharing lane>>4 group)
#pragma unroll
    for (int r = 0; r < 8; ++r) {
      const int q = q0 + wid * 16 + r + 8 * hi;
      float mx = -1e30f;
#pragma unroll
      for (int nt = 0; nt < 4; ++nt) {
        int key = kc + nt * 16 + lo;
        float sv = s[nt][r];
        int d = q - key; if (d < 0) d = -d;
        if (d > HWIN) sv = -1e30f;
        s[nt][r] = sv;
        mx = fmaxf(mx, sv);
      }
#pragma unroll
      for (int off = 1; off < 16; off <<= 1) mx = fmaxf(mx, __shfl_xor(mx, off, 32));
      float mnew = fmaxf(mrow[r], mx);
      float sc = __expf(mrow[r] - mnew);
      lrow[r] *= sc;
#pragma unroll
      for (int nt = 0; nt < 4; ++nt) acc[nt][r] = acc[nt][r] * sc;
      float psum = 0.f;
#pragma unroll
      for (int nt = 0; nt < 4; ++nt) {
        float sv = s[nt][r];
        float pv = (sv < -1e29f) ? 0.f : __expf(sv - mnew);
        s[nt][r] = pv;
        psum += pv;
      }
#pragma unroll
      for (int off = 1; off < 16; off <<= 1) psum += __shfl_xor(psum, off, 32);
      lrow[r] += psum;
      mrow[r] = mnew;
    }

    // P: C-layout -> LDS -> A-layout (per-wave region, in-order LDS)
    _Float16* pbase = &Plds[wid * 16 * 72];
#pragma unroll
    for (int nt = 0; nt < 4; ++nt)
#pragma unroll
      for (int r = 0; r < 8; ++r)
        pbase[(r + 8 * hi) * 72 + nt * 16 + lo] = (_Float16)s[nt][r];

    v16h pf[2];
#pragma unroll
    for (int st = 0; st < 2; ++st) pf[st] = load_afrag(pbase + st * 32, 72, lo, hi);

    // O += P * V
#pragma unroll
    for (int nt = 0; nt < 4; ++nt) {
#pragma unroll
      for (int st = 0; st < 2; ++st) {
        v16h bf = load_bfrag(&Vlds[(nt * 16) * 72 + st * 32], 72, lo, hi);
        acc[nt] = wmma16(pf[st], bf, acc[nt]);
      }
    }
    __syncthreads();
  }

  // normalize + write context
#pragma unroll
  for (int r = 0; r < 8; ++r) {
    const int q = q0 + wid * 16 + r + 8 * hi;
    float inv = (lrow[r] > 0.f) ? 1.f / lrow[r] : 0.f;
    float* orow = ctx + ((size_t)(b * TT + q)) * DD + h * HDIM;
#pragma unroll
    for (int nt = 0; nt < 4; ++nt) orow[nt * 16 + lo] = acc[nt][r] * inv;
  }
}

// ---------------------------------------------------------------------------
// LayerNorm: one wave per row of 512, shuffle reductions
// ---------------------------------------------------------------------------
__global__ __launch_bounds__(256) void ln_rows(
    const float* __restrict__ X, const float* __restrict__ g,
    const float* __restrict__ be, float* __restrict__ Y, int rows) {
  const int wid  = threadIdx.x >> 5;
  const int lane = threadIdx.x & 31;
  const int row  = blockIdx.x * 8 + wid;
  if (row >= rows) return;
  const float* xr = X + (size_t)row * DD;
  float v[16];
  float s = 0.f, sq = 0.f;
#pragma unroll
  for (int i = 0; i < 4; ++i) {
    float4 f = ((const float4*)xr)[lane * 4 + i];
    v[i * 4 + 0] = f.x; v[i * 4 + 1] = f.y; v[i * 4 + 2] = f.z; v[i * 4 + 3] = f.w;
    s  += f.x + f.y + f.z + f.w;
    sq += f.x * f.x + f.y * f.y + f.z * f.z + f.w * f.w;
  }
#pragma unroll
  for (int off = 1; off < 32; off <<= 1) {
    s  += __shfl_xor(s, off, 32);
    sq += __shfl_xor(sq, off, 32);
  }
  const float mean = s * (1.f / DD);
  const float var  = sq * (1.f / DD) - mean * mean;
  const float rstd = rsqrtf(var + 1e-5f);
  float* yr = Y + (size_t)row * DD;
#pragma unroll
  for (int i = 0; i < 16; ++i) {
    int col = lane * 16 + i;
    yr[col] = (v[i] - mean) * rstd * g[col] + be[col];
  }
}

// repack conv_w [O][I][3] -> [3][O][I]
__global__ void repack_conv(const float* __restrict__ cw, float* __restrict__ wt) {
  int i = blockIdx.x * 256 + threadIdx.x;
  if (i >= DD * DD * 3) return;
  int tap = i % 3;
  int rest = i / 3;
  int din = rest % DD;
  int o = rest / DD;
  wt[(size_t)tap * DD * DD + (size_t)o * DD + din] = cw[i];
}

extern "C" void kernel_launch(void* const* d_in, const int* in_sizes, int n_in,
                              void* d_out, int out_size, void* d_ws, size_t ws_size,
                              hipStream_t stream) {
  const float* x      = (const float*)d_in[0];
  // d_in[1] = pe: unused by the reference
  const float* conv_w = (const float*)d_in[2];
  const float* conv_b = (const float*)d_in[3];
  const float* in_w   = (const float*)d_in[4];
  const float* in_b   = (const float*)d_in[5];
  const float* out_w  = (const float*)d_in[6];
  const float* out_b  = (const float*)d_in[7];
  const float* ln1_g  = (const float*)d_in[8];
  const float* ln1_b  = (const float*)d_in[9];
  const float* w1     = (const float*)d_in[10];
  const float* b1     = (const float*)d_in[11];
  const float* w2     = (const float*)d_in[12];
  const float* b2     = (const float*)d_in[13];
  const float* ln2_g  = (const float*)d_in[14];
  const float* ln2_b  = (const float*)d_in[15];
  float* out = (float*)d_out;
  float* ws  = (float*)d_ws;

  const size_t SZ = (size_t)MROWS * DD;
  float* WTAP = ws;
  float* VIN  = WTAP + (size_t)3 * DD * DD;
  float* QB   = VIN + SZ;
  float* KB   = QB + SZ;
  float* VB   = KB + SZ;
  float* CTX  = VB + SZ;
  float* PRE1 = CTX + SZ;
  float* HB   = PRE1 + SZ;
  float* FF1  = HB + SZ;
  float* PRE2 = FF1 + (size_t)MROWS * 2 * DD;

  dim3 blk(256);
  repack_conv<<<dim3((3 * DD * DD + 255) / 256), blk, 0, stream>>>(conv_w, WTAP);

  // conv1d as 3-tap shifted GEMM -> v_in
  gemm_wmma<<<dim3(DD / 128, MROWS / 128), blk, 0, stream>>>(
      x, WTAP, conv_b, nullptr, VIN, MROWS, DD, DD, 0, 3);
  // Q = x Wq^T + bq ; K = x Wk^T + bk ; V = v_in Wv^T + bv
  gemm_wmma<<<dim3(DD / 128, MROWS / 128), blk, 0, stream>>>(
      x, in_w, in_b, nullptr, QB, MROWS, DD, DD, 0, 1);
  gemm_wmma<<<dim3(DD / 128, MROWS / 128), blk, 0, stream>>>(
      x, in_w + (size_t)DD * DD, in_b + DD, nullptr, KB, MROWS, DD, DD, 0, 1);
  gemm_wmma<<<dim3(DD / 128, MROWS / 128), blk, 0, stream>>>(
      VIN, in_w + (size_t)2 * DD * DD, in_b + 2 * DD, nullptr, VB, MROWS, DD, DD, 0, 1);
  // banded flash attention
  attn_wmma<<<dim3(TT / 128, 2 * NH), blk, 0, stream>>>(QB, KB, VB, CTX);
  // out_proj + residual(x) -> pre-LN1 ; LN1 -> h
  gemm_wmma<<<dim3(DD / 128, MROWS / 128), blk, 0, stream>>>(
      CTX, out_w, out_b, x, PRE1, MROWS, DD, DD, 0, 1);
  ln_rows<<<dim3(MROWS / 8), blk, 0, stream>>>(PRE1, ln1_g, ln1_b, HB, MROWS);
  // FFN: relu(h W1^T + b1) W2^T + b2 + h -> pre-LN2 ; LN2 -> out
  gemm_wmma<<<dim3((2 * DD) / 128, MROWS / 128), blk, 0, stream>>>(
      HB, w1, b1, nullptr, FF1, MROWS, 2 * DD, DD, 1, 1);
  gemm_wmma<<<dim3(DD / 128, MROWS / 128), blk, 0, stream>>>(
      FF1, w2, b2, HB, PRE2, MROWS, DD, 2 * DD, 0, 1);
  ln_rows<<<dim3(MROWS / 8), blk, 0, stream>>>(PRE2, ln2_g, ln2_b, out, MROWS);
}
